// CrossAttention_56916906607183
// MI455X (gfx1250) — matrix operand order repaired and checked by reading, not compiled, not run on hardware
//
#include <hip/hip_runtime.h>

// ---------------------------------------------------------------------------
// CDNA5 (gfx1250) wave32 WMMA cross-attention
//   B=2, Lq=1024, Lkv=2048, D=1024, H=16, hd=64
// ---------------------------------------------------------------------------

typedef __attribute__((ext_vector_type(16))) __bf16 v16bf;
typedef __attribute__((ext_vector_type(8)))  __bf16 v8bf;
typedef __attribute__((ext_vector_type(8)))  float  v8f;

union Frag16 { v16bf v; v8bf h[2]; };

__device__ __forceinline__ v8f vzero8() {
  v8f z = {0.f, 0.f, 0.f, 0.f, 0.f, 0.f, 0.f, 0.f};
  return z;
}

__device__ __forceinline__ v8f wmma_bf16(v16bf a, v16bf b, v8f c) {
  // D = A(16x32 bf16) x B(32x16 bf16) + C(16x16 f32)
  return __builtin_amdgcn_wmma_f32_16x16x32_bf16(
      /*neg_a=*/false, a, /*neg_b=*/false, b,
      /*c_mod=*/(short)0, c, /*reuse_a=*/false, /*reuse_b=*/false);
}

// A fragment (16x32, row-major X): lane lr = M row; lanes 0-15 hold K {0..7,16..23},
// lanes 16-31 hold K {8..15,24..31}  -> two contiguous 16B loads.
__device__ __forceinline__ v16bf load_afrag(const __bf16* __restrict__ row, int k0, int lh) {
  Frag16 f;
  f.h[0] = *(const v8bf*)(row + k0 + lh * 8);
  f.h[1] = *(const v8bf*)(row + k0 + 16 + lh * 8);
  return f.v;
}

// B fragment (32x16): lane lr = N col; lanes 0-15 hold K {0..7,16..23}, lanes 16-31 hold K {8..15,24..31};
// B[k][n] = row_n[k] (row-major [N,K] source) -> two contiguous 16B loads.
__device__ __forceinline__ v16bf load_bfrag(const __bf16* __restrict__ row_n, int k0, int lh) {
  Frag16 f;
  f.h[0] = *(const v8bf*)(row_n + k0 + lh * 8);
  f.h[1] = *(const v8bf*)(row_n + k0 + 16 + lh * 8);
  return f.v;
}

// ---------------------------------------------------------------------------
// elementwise prep
// ---------------------------------------------------------------------------
__global__ void zero_f32(float* __restrict__ p, int n) {
  int i = blockIdx.x * blockDim.x + threadIdx.x;
  if (i < n) p[i] = 0.f;
}

__global__ void add_emb_bf16(const float* __restrict__ x, const float* __restrict__ emb,
                             __bf16* __restrict__ o, int n) {
  int i = blockIdx.x * blockDim.x + threadIdx.x;
  if (i < n) o[i] = (__bf16)(x[i] + emb[i & 1023]);
}

__global__ void cast_bf16(const float* __restrict__ x, __bf16* __restrict__ o, int n) {
  int i = blockIdx.x * blockDim.x + threadIdx.x;
  if (i < n) o[i] = (__bf16)x[i];
}

// ---------------------------------------------------------------------------
// Generic WMMA GEMM:  C[M,1024] = X[M,1024](bf16) * W[1024,1024]^T(bf16) + bias
// One wave computes a 32(M) x 64(N) tile (2x4 register blocking): each B
// fragment is reused across two A fragments -> 12 b128 loads per 8 WMMAs.
// mode: 0 -> bf16 out [B,H,L,hd]
//       2 -> bf16 out transposed [B,H,hd,L]
//       3 -> f32 out [M,1024]
// ---------------------------------------------------------------------------
__global__ void __launch_bounds__(256)
gemm_wmma(const __bf16* __restrict__ X, const __bf16* __restrict__ W,
          const float* __restrict__ bias, void* __restrict__ outp,
          int M, int L, int mode) {
  const int lane = threadIdx.x & 31;
  const int wv   = threadIdx.x >> 5;
  const int lh   = lane >> 4;
  const int lr   = lane & 15;
  const int m0   = (blockIdx.y * 8 + wv) * 32;
  const int n0   = blockIdx.x * 64;
  if (m0 >= M) return;

  const __bf16* xrow0 = X + (size_t)(m0 + lr) * 1024;
  const __bf16* xrow1 = X + (size_t)(m0 + 16 + lr) * 1024;
  const __bf16* wrow  = W + (size_t)(n0 + lr) * 1024;

  v8f acc[2][4];
#pragma unroll
  for (int u = 0; u < 2; ++u)
#pragma unroll
    for (int t = 0; t < 4; ++t) acc[u][t] = vzero8();

  for (int k0 = 0; k0 < 1024; k0 += 32) {
    v16bf a0 = load_afrag(xrow0, k0, lh);
    v16bf a1 = load_afrag(xrow1, k0, lh);
#pragma unroll
    for (int t = 0; t < 4; ++t) {
      v16bf b = load_bfrag(wrow + (size_t)t * 16 * 1024, k0, lh);
      acc[0][t] = wmma_bf16(a0, b, acc[0][t]);
      acc[1][t] = wmma_bf16(a1, b, acc[1][t]);
    }
  }

#pragma unroll
  for (int u = 0; u < 2; ++u) {
#pragma unroll
    for (int t = 0; t < 4; ++t) {
#pragma unroll
      for (int r = 0; r < 8; ++r) {
        const int m = m0 + u * 16 + r + 8 * lh;  // output row
        const int n = n0 + t * 16 + lr;          // output col
        const float v = acc[u][t][r] + bias[n];
        if (mode == 3) {
          ((float*)outp)[(size_t)m * 1024 + n] = v;
        } else {
          const int bb = m / L, mm = m % L;
          const int hh = n >> 6, dd = n & 63;
          size_t idx;
          if (mode == 2)  // V stored transposed: [B,H,hd,L]
            idx = (((size_t)bb * 16 + hh) * 64 + dd) * (size_t)L + mm;
          else            // Q/K: [B,H,L,hd]
            idx = (((size_t)bb * 16 + hh) * (size_t)L + mm) * 64 + dd;
          ((__bf16*)outp)[idx] = (__bf16)v;
        }
      }
    }
  }
}

// ---------------------------------------------------------------------------
// Attention: one wave per (b, h, 16-row q tile). Two-pass softmax with score
// recompute; P tile bounced through LDS to reach WMMA A-layout; head-averaged
// attention weights accumulated with global f32 atomics.
// ---------------------------------------------------------------------------
__global__ void __launch_bounds__(256)
attn_kernel(const __bf16* __restrict__ Q,   // [B,H,1024,64]
            const __bf16* __restrict__ K,   // [B,H,2048,64]
            const __bf16* __restrict__ VT,  // [B,H,64,2048]
            __bf16* __restrict__ ctx,       // [B*1024, 1024]
            float* __restrict__ attn_out) { // [B,1024,2048] (pre-zeroed)
  __shared__ __bf16 plds[8][16][32];        // per-wave P staging (8 KB)

  const int lane = threadIdx.x & 31;
  const int wv   = threadIdx.x >> 5;
  const int lh   = lane >> 4;
  const int lr   = lane & 15;
  const int gw   = blockIdx.x * 8 + wv;     // 2048 waves total
  const int qt   = gw & 63;
  const int h    = (gw >> 6) & 15;
  const int b    = gw >> 10;

  const __bf16* Qh = Q  + (((size_t)(b * 16 + h)) * 1024 + qt * 16) * 64;
  const __bf16* Kh = K  + ((size_t)(b * 16 + h)) * 2048 * 64;
  const __bf16* Vh = VT + ((size_t)(b * 16 + h)) * 64 * 2048;

  // Q tile (16x64) as two A fragments (K = 0..31, 32..63), resident in VGPRs
  const __bf16* qrow = Qh + (size_t)lr * 64;
  const v16bf aq0 = load_afrag(qrow, 0, lh);
  const v16bf aq1 = load_afrag(qrow, 32, lh);

  float Mr[8], Lr[8];
#pragma unroll
  for (int r = 0; r < 8; ++r) { Mr[r] = -1e30f; Lr[r] = 0.f; }

  const __bf16* krow = Kh + (size_t)lr * 64;

  // ---- pass 1: streaming row max / row sum ----
  for (int kt = 0; kt < 128; ++kt) {
    const __bf16* kr = krow + (size_t)kt * 16 * 64;
    v8f s = vzero8();
    s = wmma_bf16(aq0, load_bfrag(kr, 0, lh), s);
    s = wmma_bf16(aq1, load_bfrag(kr, 32, lh), s);
#pragma unroll
    for (int r = 0; r < 8; ++r) {
      float v = s[r] * 0.125f;               // 1/sqrt(64)
      float mx = v;
#pragma unroll
      for (int off = 1; off < 16; off <<= 1) mx = fmaxf(mx, __shfl_xor(mx, off, 16));
      const float nm = fmaxf(Mr[r], mx);
      float p = __expf(v - nm);
#pragma unroll
      for (int off = 1; off < 16; off <<= 1) p += __shfl_xor(p, off, 16);
      Lr[r] = Lr[r] * __expf(Mr[r] - nm) + p;
      Mr[r] = nm;
    }
  }

  float invL[8];
#pragma unroll
  for (int r = 0; r < 8; ++r) invL[r] = (Lr[r] > 0.f) ? 1.f / Lr[r] : 0.f;

  // ---- pass 2: exact probabilities, attn-weight output, ctx = P x V ----
  v8f o[4];
#pragma unroll
  for (int t = 0; t < 4; ++t) o[t] = vzero8();

  float* attn_base = attn_out + ((size_t)b * 1024 + qt * 16) * 2048;

  for (int kp = 0; kp < 64; ++kp) {          // 32 kv rows per iteration
#pragma unroll
    for (int jj = 0; jj < 2; ++jj) {
      const int kt = kp * 2 + jj;
      const __bf16* kr = krow + (size_t)kt * 16 * 64;
      v8f s = vzero8();
      s = wmma_bf16(aq0, load_bfrag(kr, 0, lh), s);
      s = wmma_bf16(aq1, load_bfrag(kr, 32, lh), s);
#pragma unroll
      for (int r = 0; r < 8; ++r) {
        const float p = __expf(s[r] * 0.125f - Mr[r]) * invL[r];
        plds[wv][r + 8 * lh][jj * 16 + lr] = (__bf16)p;
        atomicAdd(attn_base + (size_t)(r + 8 * lh) * 2048 + kt * 16 + lr,
                  p * 0.0625f);              // mean over 16 heads
      }
    }
    // LDS stores above feed the cross-lane A-fragment loads below
    __asm__ volatile("s_wait_dscnt 0" ::: "memory");

    Frag16 ap;
    const __bf16* prow = &plds[wv][lr][0];
    ap.h[0] = *(const v8bf*)(prow + lh * 8);
    ap.h[1] = *(const v8bf*)(prow + 16 + lh * 8);

#pragma unroll
    for (int t = 0; t < 4; ++t) {
      const __bf16* vrow = Vh + (size_t)(t * 16 + lr) * 2048 + kp * 32;
      Frag16 bv;
      bv.h[0] = *(const v8bf*)(vrow + lh * 8);
      bv.h[1] = *(const v8bf*)(vrow + 16 + lh * 8);
      o[t] = wmma_bf16(ap.v, bv.v, o[t]);
    }
  }

  __bf16* crow = ctx + ((size_t)b * 1024 + qt * 16) * 1024 + h * 64;
#pragma unroll
  for (int t = 0; t < 4; ++t)
#pragma unroll
    for (int r = 0; r < 8; ++r)
      crow[(size_t)(r + 8 * lh) * 1024 + t * 16 + lr] = (__bf16)o[t][r];
}

// ---------------------------------------------------------------------------
// host launcher
// ---------------------------------------------------------------------------
extern "C" void kernel_launch(void* const* d_in, const int* in_sizes, int n_in,
                              void* d_out, int out_size, void* d_ws, size_t ws_size,
                              hipStream_t stream) {
  (void)in_sizes; (void)n_in; (void)out_size; (void)ws_size;

  const float* x_cond    = (const float*)d_in[0];  // [2,2048,1024]
  const float* x_query   = (const float*)d_in[1];  // [2,1024,1024]
  const float* tok_emb   = (const float*)d_in[2];  // [2,1024]
  const float* in_proj_w = (const float*)d_in[3];  // [3072,1024]
  const float* in_proj_b = (const float*)d_in[4];  // [3072]
  const float* out_w     = (const float*)d_in[5];  // [1024,1024]
  const float* out_b     = (const float*)d_in[6];  // [1024]

  float* outF  = (float*)d_out;                     // fused_output [2,1024,1024]
  float* attnF = outF + (size_t)2 * 1024 * 1024;    // attn_weights [2,1024,2048]

  char* p = (char*)d_ws;
  __bf16* q_in  = (__bf16*)p; p += (size_t)2097152 * 2;  // [2*1024,1024]
  __bf16* k_in  = (__bf16*)p; p += (size_t)4194304 * 2;  // [2*2048,1024]
  __bf16* w_qkv = (__bf16*)p; p += (size_t)3145728 * 2;  // [3072,1024]
  __bf16* w_out = (__bf16*)p; p += (size_t)1048576 * 2;  // [1024,1024]
  __bf16* Qb    = (__bf16*)p; p += (size_t)2097152 * 2;  // [2,16,1024,64]
  __bf16* Kb    = (__bf16*)p; p += (size_t)4194304 * 2;  // [2,16,2048,64]
  __bf16* VTb   = (__bf16*)p; p += (size_t)4194304 * 2;  // [2,16,64,2048]
  __bf16* ctxb  = (__bf16*)p; p += (size_t)2097152 * 2;  // [2*1024,1024]

  // zero the attention-weight accumulator region
  zero_f32<<<dim3(4194304 / 256), 256, 0, stream>>>(attnF, 4194304);

  // token-type embedding add + bf16 cast
  add_emb_bf16<<<dim3(2097152 / 256), 256, 0, stream>>>(x_query, tok_emb + 1024, q_in, 2097152);
  add_emb_bf16<<<dim3(4194304 / 256), 256, 0, stream>>>(x_cond,  tok_emb,        k_in, 4194304);
  cast_bf16<<<dim3(3145728 / 256), 256, 0, stream>>>(in_proj_w, w_qkv, 3145728);
  cast_bf16<<<dim3(1048576 / 256), 256, 0, stream>>>(out_w,     w_out, 1048576);

  // QKV projections (WMMA bf16 GEMMs); each wave does a 32x64 tile
  gemm_wmma<<<dim3(16,  8), 256, 0, stream>>>(q_in, w_qkv,                   in_proj_b,        (void*)Qb,  2048, 1024, 0);
  gemm_wmma<<<dim3(16, 16), 256, 0, stream>>>(k_in, w_qkv + (size_t)1048576, in_proj_b + 1024, (void*)Kb,  4096, 2048, 0);
  gemm_wmma<<<dim3(16, 16), 256, 0, stream>>>(k_in, w_qkv + (size_t)2097152, in_proj_b + 2048, (void*)VTb, 4096, 2048, 2);

  // attention (scores + softmax + ctx, WMMA)
  attn_kernel<<<dim3(256), 256, 0, stream>>>(Qb, Kb, VTb, ctxb, attnF);

  // output projection -> f32
  gemm_wmma<<<dim3(16,  8), 256, 0, stream>>>(ctxb, w_out, out_b, (void*)outF, 2048, 1024, 3);
}
